// SpFAContrast_20641612824896
// MI455X (gfx1250) — matrix operand order repaired and checked by reading, not compiled, not run on hardware
//
#include <hip/hip_runtime.h>
#include <hip/hip_bf16.h>
#include <math.h>

// ---------------- problem constants ----------------
#define NPTS   524288
#define SSEG   256
#define SEGSZ  2048
#define CH     96
#define KS     30
#define SK     7680          // SSEG*KS
#define NEGS   30
#define TINV   2.0f          // 1/0.5
#define EPSF   1e-7f
#define WPAD   104           // LDS row pad for weights (208B rows, 16B aligned)

typedef __bf16 bf16;
typedef __attribute__((ext_vector_type(16))) __bf16 bf16x16;
typedef __attribute__((ext_vector_type(8)))  float  f32x8;

// Native conversion: lowers to hardware bf16 convert (RNE) on gfx1250.
__device__ __forceinline__ bf16 f2bf(float f) { return (bf16)f; }

__device__ __forceinline__ f32x8 zero8() {
    f32x8 z;
#pragma unroll
    for (int i = 0; i < 8; ++i) z[i] = 0.0f;
    return z;
}

// A fragment: 16x32 bf16 tile of x rows (row-major, ld=96).
__device__ __forceinline__ bf16x16 load_a_f32(const float* __restrict__ p, int kt, int kh) {
    bf16x16 a;
    int k0 = kt + kh * 16;
    int k1 = kt + kh * 16 + 8;
#pragma unroll
    for (int e = 0; e < 8; ++e) {
        a[e]     = f2bf(p[k0 + e]);
        a[e + 8] = f2bf(p[k1 + e]);
    }
    return a;
}

// B fragment from LDS-staged weight W[n][k] (rows padded to WPAD):
// B[k,n] = W[n][k]; lane holds n=(lane&15), K = kt + (lane>>4)*16 + e
__device__ __forceinline__ bf16x16 load_b_lds(const bf16* __restrict__ sW, int nBase,
                                              int kt, int lane) {
    int n  = nBase + (lane & 15);
    int kh = lane >> 4;
    const bf16* p = sW + n * WPAD + kt + kh * 16;
    bf16x16 b;
#pragma unroll
    for (int e = 0; e < 16; ++e) b[e] = p[e];
    return b;
}

// B fragment straight from f32 row-major source (ld=96): B[k,n] = src[n][k]
__device__ __forceinline__ bf16x16 load_b_f32(const float* __restrict__ src, int nBase,
                                              int kt, int lane) {
    int n  = nBase + (lane & 15);
    int kh = lane >> 4;
    const float* p = src + (size_t)n * CH + kt + kh * 16;
    bf16x16 b;
#pragma unroll
    for (int e = 0; e < 16; ++e) b[e] = f2bf(p[e]);
    return b;
}

#define WMMA_BF16(A, B, C) \
    __builtin_amdgcn_wmma_f32_16x16x32_bf16(false, (A), false, (B), (short)0, (C), false, false)

// Stage 96x96 bf16 weights (global, row stride 96) into LDS rows padded to WPAD,
// using CDNA5 async global->LDS 128-bit copies tracked by ASYNCcnt.
__device__ __forceinline__ void stage_weights_async(const bf16* __restrict__ wT,
                                                    bf16* __restrict__ sW,
                                                    int tid, int nthreads) {
    for (int c = tid; c < CH * 12; c += nthreads) {      // 12 x 16B chunks per row
        int n = c / 12, ck = c % 12;
        unsigned ldsOff = (unsigned)(size_t)(sW + n * WPAD) + ck * 16;
        const void* g = (const char*)(wT + (size_t)n * CH) + ck * 16;
        asm volatile("global_load_async_to_lds_b128 %0, %1, off"
                     :: "v"(ldsOff), "v"(g) : "memory");
    }
    asm volatile("s_wait_asynccnt 0x0" ::: "memory");
}

// ---------------- K0: convert both weight matrices to bf16 ----------------
__global__ __launch_bounds__(256) void convw_kernel(const float* __restrict__ wq,
                                                    const float* __restrict__ wk,
                                                    bf16* __restrict__ bq,
                                                    bf16* __restrict__ bk) {
    int i = blockIdx.x * 256 + threadIdx.x;
    if (i < CH * CH) {
        bq[i] = f2bf(wq[i]);
        bk[i] = f2bf(wk[i]);
    }
}

// ---------------- P1: GEMM y = x@W^T + b, accumulate BN stats (+opt raw seg sums) ----
// 256 threads = 8 waves, 32 rows/wave -> 256 rows/block.
template <bool SEGSUM>
__global__ __launch_bounds__(256) void stats_kernel(const float* __restrict__ x,
                                                    const bf16* __restrict__ wT,
                                                    const float* __restrict__ bias,
                                                    float* __restrict__ statSum,
                                                    float* __restrict__ statSq,
                                                    float* __restrict__ segSum) {
    __shared__ __align__(16) bf16 sW[CH * WPAD];
    __shared__ float sSum[CH], sSq[CH];
    int tid = threadIdx.x;
    stage_weights_async(wT, sW, tid, 256);
    if (tid < CH) { sSum[tid] = 0.f; sSq[tid] = 0.f; }
    __syncthreads();

    int wave = tid >> 5, lane = tid & 31;
    int m = lane & 15, kh = lane >> 4;
    size_t rowBase = (size_t)blockIdx.x * 256 + wave * 32;
    const float* prow0 = x + (rowBase + m) * CH;
    const float* prow1 = prow0 + 16 * CH;

    f32x8 acc[12];
#pragma unroll
    for (int t = 0; t < 12; ++t) acc[t] = zero8();

#pragma unroll
    for (int kt = 0; kt < CH; kt += 32) {
        bf16x16 a0 = load_a_f32(prow0, kt, kh);
        bf16x16 a1 = load_a_f32(prow1, kt, kh);
        bf16x16 b[6];
#pragma unroll
        for (int nt = 0; nt < 6; ++nt) b[nt] = load_b_lds(sW, nt * 16, kt, lane);
#pragma unroll
        for (int nt = 0; nt < 6; ++nt) acc[nt] = WMMA_BF16(a0, b[nt], acc[nt]);
#pragma unroll
        for (int nt = 0; nt < 6; ++nt) acc[6 + nt] = WMMA_BF16(a1, b[nt], acc[6 + nt]);
    }

    int n15 = lane & 15;
#pragma unroll
    for (int nt = 0; nt < 6; ++nt) {
        int n = nt * 16 + n15;
        float bb = bias[n];
        float s = 0.f, q = 0.f;
#pragma unroll
        for (int g = 0; g < 2; ++g)
#pragma unroll
            for (int r = 0; r < 8; ++r) {
                float v = acc[g * 6 + nt][r] + bb;
                s += v; q += v * v;
            }
        atomicAdd(&sSum[n], s);
        atomicAdd(&sSq[n], q);
    }

    if (SEGSUM && tid < CH) {          // raw view2 segment sums (256 contiguous rows/blk)
        float part = 0.f;
        const float* base = x + (size_t)blockIdx.x * 256 * CH + tid;
        for (int rr = 0; rr < 256; ++rr) part += base[(size_t)rr * CH];
        int seg = (blockIdx.x * 256) >> 11;
        atomicAdd(&segSum[seg * CH + tid], part);
    }

    __syncthreads();
    if (tid < CH) {
        atomicAdd(&statSum[tid], sSum[tid]);
        atomicAdd(&statSq[tid],  sSq[tid]);
    }
}

// ---------------- K3: finalize BN affine (scale/shift fold bias) ----------------
__global__ __launch_bounds__(256) void bn_finalize_kernel(
    const float* sumQ, const float* sqQ, const float* sumK, const float* sqK,
    const float* bq, const float* gq, const float* beq,
    const float* bk, const float* gk, const float* bek,
    float* aQ, float* cQ, float* aK, float* cK) {
    int t = threadIdx.x;
    if (t < 2 * CH) {
        int h = t / CH, c = t % CH;
        const float* s  = h ? sumK : sumQ;
        const float* s2 = h ? sqK  : sqQ;
        const float* bb = h ? bk   : bq;
        const float* gg = h ? gk   : gq;
        const float* be = h ? bek  : beq;
        float mu  = s[c] / (float)NPTS;
        float var = s2[c] / (float)NPTS - mu * mu;
        float sc  = gg[c] * rsqrtf(var + 1e-5f);
        float sh  = be[c] + (bb[c] - mu) * sc;
        (h ? aK : aQ)[c] = sc;
        (h ? cK : cQ)[c] = sh;
    }
}

// ---------------- P2: view2 GEMM again, BN+ReLU, segment sums ----------------
__global__ __launch_bounds__(256) void pass2_kernel(const float* __restrict__ x,
                                                    const bf16* __restrict__ wT,
                                                    const float* __restrict__ bnA,
                                                    const float* __restrict__ bnC,
                                                    float* __restrict__ segG) {
    __shared__ __align__(16) bf16 sW[CH * WPAD];
    __shared__ float sSeg[CH];
    int tid = threadIdx.x;
    stage_weights_async(wT, sW, tid, 256);
    if (tid < CH) sSeg[tid] = 0.f;
    __syncthreads();

    int wave = tid >> 5, lane = tid & 31;
    int m = lane & 15, kh = lane >> 4;
    size_t rowBase = (size_t)blockIdx.x * 256 + wave * 32;
    const float* prow0 = x + (rowBase + m) * CH;
    const float* prow1 = prow0 + 16 * CH;

    f32x8 acc[12];
#pragma unroll
    for (int t = 0; t < 12; ++t) acc[t] = zero8();
#pragma unroll
    for (int kt = 0; kt < CH; kt += 32) {
        bf16x16 a0 = load_a_f32(prow0, kt, kh);
        bf16x16 a1 = load_a_f32(prow1, kt, kh);
        bf16x16 b[6];
#pragma unroll
        for (int nt = 0; nt < 6; ++nt) b[nt] = load_b_lds(sW, nt * 16, kt, lane);
#pragma unroll
        for (int nt = 0; nt < 6; ++nt) acc[nt] = WMMA_BF16(a0, b[nt], acc[nt]);
#pragma unroll
        for (int nt = 0; nt < 6; ++nt) acc[6 + nt] = WMMA_BF16(a1, b[nt], acc[6 + nt]);
    }

    int n15 = lane & 15;
#pragma unroll
    for (int nt = 0; nt < 6; ++nt) {
        int n = nt * 16 + n15;
        float sc = bnA[n], sh = bnC[n];
        float s = 0.f;
#pragma unroll
        for (int g = 0; g < 2; ++g)
#pragma unroll
            for (int r = 0; r < 8; ++r)
                s += fmaxf(acc[g * 6 + nt][r] * sc + sh, 0.f);
        atomicAdd(&sSeg[n], s);
    }
    __syncthreads();
    if (tid < CH) {
        int seg = (blockIdx.x * 256) >> 11;
        atomicAdd(&segG[seg * CH + tid], sSeg[tid]);
    }
}

// ---------------- K5: normalize segment means -> anchors ----------------
__global__ __launch_bounds__(128) void anchor_kernel(const float* __restrict__ segR,
                                                     const float* __restrict__ segG,
                                                     float* __restrict__ aR,
                                                     float* __restrict__ a2) {
    int r = blockIdx.x, tid = threadIdx.x;
    __shared__ float red[128];
    for (int which = 0; which < 2; ++which) {
        const float* src = which ? segG : segR;
        float* dst       = which ? a2   : aR;
        float mval = (tid < CH) ? src[r * CH + tid] * (1.0f / (float)SEGSZ) : 0.f;
        red[tid] = mval * mval;
        __syncthreads();
        for (int s = 64; s > 0; s >>= 1) {
            if (tid < s) red[tid] += red[tid + s];
            __syncthreads();
        }
        float rn = 1.0f / (sqrtf(red[0]) + EPSF);
        __syncthreads();
        if (tid < CH) dst[r * CH + tid] = mval * rn;
        __syncthreads();
    }
}

// ---------------- P3: sampled rows: raw l2norm + gk head GEMM+BN+ReLU+l2norm ----
__global__ __launch_bounds__(128) void samples_kernel(const float* __restrict__ v1,
                                                      const int* __restrict__ idx,
                                                      const bf16* __restrict__ wT,
                                                      const float* __restrict__ bnA,
                                                      const float* __restrict__ bnC,
                                                      float* __restrict__ sRaw,
                                                      float* __restrict__ s2) {
    __shared__ __align__(16) bf16 sW[CH * WPAD];
    __shared__ float sY[4 * 16 * 100];
    int tid = threadIdx.x;
    stage_weights_async(wT, sW, tid, 128);
    __syncthreads();

    int wave = tid >> 5, lane = tid & 31;
    int m = lane & 15, kh = lane >> 4;
    int sIdx = blockIdx.x * 64 + wave * 16 + m;
    const float* prow = v1 + (size_t)idx[sIdx] * CH;

    f32x8 acc[6];
#pragma unroll
    for (int t = 0; t < 6; ++t) acc[t] = zero8();
#pragma unroll
    for (int kt = 0; kt < CH; kt += 32) {
        bf16x16 a = load_a_f32(prow, kt, kh);
        bf16x16 b[6];
#pragma unroll
        for (int nt = 0; nt < 6; ++nt) b[nt] = load_b_lds(sW, nt * 16, kt, lane);
#pragma unroll
        for (int nt = 0; nt < 6; ++nt) acc[nt] = WMMA_BF16(a, b[nt], acc[nt]);
    }

    int n15 = lane & 15;
    float* my = sY + wave * 1600;
#pragma unroll
    for (int nt = 0; nt < 6; ++nt) {
        int n = nt * 16 + n15;
        float sc = bnA[n], sh = bnC[n];
#pragma unroll
        for (int r = 0; r < 8; ++r)
            my[(r + 8 * kh) * 100 + n] = fmaxf(acc[nt][r] * sc + sh, 0.f);
    }
    __syncthreads();

    if (tid < 64) {                      // projected samples
        const float* yrow = sY + (tid / 16) * 1600 + (tid % 16) * 100;
        float ss = 0.f;
        for (int k = 0; k < CH; ++k) ss += yrow[k] * yrow[k];
        float rn = 1.0f / (sqrtf(ss) + EPSF);
        float* dst = s2 + (size_t)(blockIdx.x * 64 + tid) * CH;
        for (int k = 0; k < CH; ++k) dst[k] = yrow[k] * rn;
    } else {                             // raw samples
        int g = blockIdx.x * 64 + (tid - 64);
        const float* src = v1 + (size_t)idx[g] * CH;
        float ss = 0.f;
        for (int k = 0; k < CH; ++k) { float v = src[k]; ss += v * v; }
        float rn = 1.0f / (sqrtf(ss) + EPSF);
        float* dst = sRaw + (size_t)g * CH;
        for (int k = 0; k < CH; ++k) dst[k] = src[k] * rn;
    }
}

// ---------------- P4: sim2 = anchor2 @ samples2^T (WMMA) ----------------
__global__ __launch_bounds__(256) void sim_kernel(const float* __restrict__ anchor2,
                                                  const float* __restrict__ samples2,
                                                  float* __restrict__ sim2) {
    int tid = threadIdx.x;
    int wave = tid >> 5, lane = tid & 31;
    int wid = blockIdx.x * 8 + wave;       // 1280 waves total
    int rowTile = wid / 80;                // 16 row tiles
    int colGrp  = wid % 80;                // 80 groups of 96 cols
    int m = lane & 15, kh = lane >> 4, n15 = lane & 15;
    const float* prow = anchor2 + (size_t)(rowTile * 16 + m) * CH;
    int colBase = colGrp * 96;

    f32x8 acc[6];
#pragma unroll
    for (int t = 0; t < 6; ++t) acc[t] = zero8();
#pragma unroll
    for (int kt = 0; kt < CH; kt += 32) {
        bf16x16 a = load_a_f32(prow, kt, kh);
        bf16x16 b[6];
#pragma unroll
        for (int nt = 0; nt < 6; ++nt) b[nt] = load_b_f32(samples2, colBase + nt * 16, kt, lane);
#pragma unroll
        for (int nt = 0; nt < 6; ++nt) acc[nt] = WMMA_BF16(a, b[nt], acc[nt]);
    }
#pragma unroll
    for (int nt = 0; nt < 6; ++nt) {
#pragma unroll
        for (int r = 0; r < 8; ++r) {
            int row = rowTile * 16 + r + 8 * kh;
            int col = colBase + nt * 16 + n15;
            sim2[(size_t)row * SK + col] = acc[nt][r];
        }
    }
}

// ---------------- P5: iterative top-30 per row ----------------
__global__ __launch_bounds__(256) void topk_kernel(const float* __restrict__ sim2,
                                                   int* __restrict__ topk,
                                                   unsigned char* __restrict__ mask) {
    int r = blockIdx.x, tid = threadIdx.x;
    const float* row = sim2 + (size_t)r * SK;
    __shared__ unsigned char flag[SK];
    __shared__ float rv[256];
    __shared__ int   ri[256];
    for (int i = tid; i < SK; i += 256) flag[i] = 0;
    __syncthreads();
    for (int it = 0; it < KS; ++it) {
        float best = -3.4e38f; int bi = SK;
        for (int c = tid; c < SK; c += 256) {
            if (!flag[c]) {
                float v = row[c];
                if (v > best || (v == best && c < bi)) { best = v; bi = c; }
            }
        }
        rv[tid] = best; ri[tid] = bi;
        __syncthreads();
        for (int s = 128; s > 0; s >>= 1) {
            if (tid < s) {
                if (rv[tid + s] > rv[tid] ||
                    (rv[tid + s] == rv[tid] && ri[tid + s] < ri[tid])) {
                    rv[tid] = rv[tid + s]; ri[tid] = ri[tid + s];
                }
            }
            __syncthreads();
        }
        if (tid == 0) {
            int b = ri[0];
            flag[b] = 1;
            topk[r * KS + it] = b;
            mask[(size_t)r * SK + b] = 1;
        }
        __syncthreads();
    }
}

// ---------------- P6a: Lcon (60 analytic columns per row) ----------------
__global__ __launch_bounds__(64) void lcon1_kernel(const float* __restrict__ aRaw,
                                                   const float* __restrict__ sRaw,
                                                   float* __restrict__ acc) {
    int r = blockIdx.x, tid = threadIdx.x;
    __shared__ float sa[CH];
    __shared__ float red[64];
    for (int i = tid; i < CH; i += 64) sa[i] = aRaw[r * CH + i];
    __syncthreads();

    float logit = -1e30f;
    if (tid < 60) {
        int col;
        if (tid < KS) col = r * KS + tid;
        else {
            int j = tid - KS;
            int p = (int)(((float)(j * (SK - KS - 1))) / (float)(NEGS - 1));
            col = (p < r * KS) ? p : p + KS;
        }
        const float* srow = sRaw + (size_t)col * CH;
        float d = 0.f;
        for (int k = 0; k < CH; ++k) d += sa[k] * srow[k];
        logit = d * TINV;
    }
    red[tid] = logit;
    __syncthreads();
    for (int s = 32; s > 0; s >>= 1) {
        if (tid < s) red[tid] = fmaxf(red[tid], red[tid + s]);
        __syncthreads();
    }
    float mx = red[0];
    __syncthreads();
    red[tid] = (tid < 60) ? __expf(logit - mx) : 0.f;
    __syncthreads();
    for (int s = 32; s > 0; s >>= 1) {
        if (tid < s) red[tid] += red[tid + s];
        __syncthreads();
    }
    float lse = mx + logf(red[0]);
    __syncthreads();
    red[tid] = (tid < KS) ? logit : 0.f;
    __syncthreads();
    for (int s = 32; s > 0; s >>= 1) {
        if (tid < s) red[tid] += red[tid + s];
        __syncthreads();
    }
    if (tid == 0) atomicAdd(&acc[0], red[0] / (float)KS - lse);
}

// ---------------- P6b: Lcon2 row scan ----------------
__global__ __launch_bounds__(256) void lcon2_kernel(const float* __restrict__ sim2,
                                                    const int* __restrict__ topk,
                                                    const unsigned char* __restrict__ mask,
                                                    const unsigned char* __restrict__ bg,
                                                    float* __restrict__ acc) {
    int r = blockIdx.x, tid = threadIdx.x;
    __shared__ float red[256];
    __shared__ unsigned char sbg[SSEG];
    for (int i = tid; i < SSEG; i += 256) sbg[i] = bg[i];
    __syncthreads();
    int bgr = sbg[r];
    const float* row = sim2 + (size_t)r * SK;
    const unsigned char* mrow = mask + (size_t)r * SK;

    float mx = -1e30f;
    for (int c = tid; c < SK; c += 256) {
        int seg = c / KS;
        bool neg = (bgr ? !sbg[seg] : (sbg[seg] != 0)) && (seg != r);
        if (neg || mrow[c]) mx = fmaxf(mx, row[c] * TINV);
    }
    red[tid] = mx;
    __syncthreads();
    for (int s = 128; s > 0; s >>= 1) {
        if (tid < s) red[tid] = fmaxf(red[tid], red[tid + s]);
        __syncthreads();
    }
    float M = red[0];
    __syncthreads();

    float sm = 0.f;
    for (int c = tid; c < SK; c += 256) {
        int seg = c / KS;
        bool neg = (bgr ? !sbg[seg] : (sbg[seg] != 0)) && (seg != r);
        if (neg || mrow[c]) sm += __expf(row[c] * TINV - M);
    }
    red[tid] = sm;
    __syncthreads();
    for (int s = 128; s > 0; s >>= 1) {
        if (tid < s) red[tid] += red[tid + s];
        __syncthreads();
    }
    float lse = M + logf(red[0]);
    __syncthreads();

    red[tid] = (tid < KS) ? row[topk[r * KS + tid]] * TINV : 0.f;
    __syncthreads();
    for (int s = 128; s > 0; s >>= 1) {
        if (tid < s) red[tid] += red[tid + s];
        __syncthreads();
    }
    if (tid == 0) atomicAdd(&acc[1], red[0] / (float)KS - lse);
}

// ---------------- K7: combine ----------------
__global__ void finalize_kernel(const float* __restrict__ acc, float* __restrict__ out) {
    if (threadIdx.x == 0 && blockIdx.x == 0)
        out[0] = -(acc[0] + acc[1]) / (float)SSEG;
}

// ---------------- workspace layout (bytes) ----------------
static constexpr size_t OFF_WQ   = 0;                         // 9216 bf16
static constexpr size_t OFF_WK   = OFF_WQ   + 18432;
static constexpr size_t OFF_STAT = OFF_WK   + 18432;          // 4*96 f32
static constexpr size_t OFF_BN   = OFF_STAT + 1536;           // 4*96 f32
static constexpr size_t OFF_SEGR = OFF_BN   + 1536;           // 256*96 f32
static constexpr size_t OFF_SEGG = OFF_SEGR + 98304;
static constexpr size_t OFF_AR   = OFF_SEGG + 98304;
static constexpr size_t OFF_A2   = OFF_AR   + 98304;
static constexpr size_t OFF_SRAW = OFF_A2   + 98304;          // 7680*96 f32
static constexpr size_t OFF_S2   = OFF_SRAW + 2949120;
static constexpr size_t OFF_SIM2 = OFF_S2   + 2949120;        // 256*7680 f32
static constexpr size_t OFF_TOPK = OFF_SIM2 + 7864320;        // 256*30 i32
static constexpr size_t OFF_MASK = OFF_TOPK + 30720;          // 256*7680 u8
static constexpr size_t OFF_ACC  = OFF_MASK + 1966080;        // 2 f32

extern "C" void kernel_launch(void* const* d_in, const int* in_sizes, int n_in,
                              void* d_out, int out_size, void* d_ws, size_t ws_size,
                              hipStream_t stream) {
    (void)in_sizes; (void)n_in; (void)out_size; (void)ws_size;
    const float* v1   = (const float*)d_in[0];
    const float* v2   = (const float*)d_in[1];
    const float* gqw  = (const float*)d_in[2];
    const float* gqb  = (const float*)d_in[3];
    const float* gqg  = (const float*)d_in[4];
    const float* gqbe = (const float*)d_in[5];
    const float* gkw  = (const float*)d_in[6];
    const float* gkb  = (const float*)d_in[7];
    const float* gkg  = (const float*)d_in[8];
    const float* gkbe = (const float*)d_in[9];
    const int*   idx  = (const int*)d_in[12];
    const unsigned char* bg = (const unsigned char*)d_in[13];

    char* ws = (char*)d_ws;
    bf16*  wqB  = (bf16*)(ws + OFF_WQ);
    bf16*  wkB  = (bf16*)(ws + OFF_WK);
    float* sumQ = (float*)(ws + OFF_STAT);
    float* sqQ  = sumQ + CH;
    float* sumK = sumQ + 2 * CH;
    float* sqK  = sumQ + 3 * CH;
    float* aQ   = (float*)(ws + OFF_BN);
    float* cQ   = aQ + CH;
    float* aK   = aQ + 2 * CH;
    float* cK   = aQ + 3 * CH;
    float* segR = (float*)(ws + OFF_SEGR);
    float* segG = (float*)(ws + OFF_SEGG);
    float* aR   = (float*)(ws + OFF_AR);
    float* a2   = (float*)(ws + OFF_A2);
    float* sRaw = (float*)(ws + OFF_SRAW);
    float* s2   = (float*)(ws + OFF_S2);
    float* sim2 = (float*)(ws + OFF_SIM2);
    int*   topk = (int*)(ws + OFF_TOPK);
    unsigned char* maskp = (unsigned char*)(ws + OFF_MASK);
    float* accp = (float*)(ws + OFF_ACC);

    hipMemsetAsync(ws + OFF_STAT, 0, 1536, stream);
    hipMemsetAsync(ws + OFF_SEGR, 0, 2 * 98304, stream);
    hipMemsetAsync(ws + OFF_MASK, 0, 1966080, stream);
    hipMemsetAsync(ws + OFF_ACC,  0, 8, stream);

    convw_kernel<<<36, 256, 0, stream>>>(gqw, gkw, wqB, wkB);
    stats_kernel<true><<<NPTS / 256, 256, 0, stream>>>(v2, wqB, gqb, sumQ, sqQ, segR);
    stats_kernel<false><<<NPTS / 256, 256, 0, stream>>>(v1, wkB, gkb, sumK, sqK, nullptr);
    bn_finalize_kernel<<<1, 256, 0, stream>>>(sumQ, sqQ, sumK, sqK,
                                              gqb, gqg, gqbe, gkb, gkg, gkbe,
                                              aQ, cQ, aK, cK);
    pass2_kernel<<<NPTS / 256, 256, 0, stream>>>(v2, wqB, aQ, cQ, segG);
    anchor_kernel<<<SSEG, 128, 0, stream>>>(segR, segG, aR, a2);
    samples_kernel<<<SK / 64, 128, 0, stream>>>(v1, idx, wkB, aK, cK, sRaw, s2);
    sim_kernel<<<160, 256, 0, stream>>>(a2, s2, sim2);
    topk_kernel<<<SSEG, 256, 0, stream>>>(sim2, topk, maskp);
    lcon1_kernel<<<SSEG, 64, 0, stream>>>(aR, sRaw, accp);
    lcon2_kernel<<<SSEG, 256, 0, stream>>>(sim2, topk, maskp, bg, accp);
    finalize_kernel<<<1, 32, 0, stream>>>(accp, (float*)d_out);
}
